// CMIA_70806830842503
// MI455X (gfx1250) — compile-verified, matrix operand order checked by reading, not run on hardware
//
#include <hip/hip_runtime.h>
#include <hip/hip_bf16.h>

typedef __attribute__((ext_vector_type(16))) _Float16 v16h;
typedef __attribute__((ext_vector_type(8)))  _Float16 v8h;
typedef __attribute__((ext_vector_type(8)))  float    v8f;
typedef __attribute__((ext_vector_type(4)))  unsigned int u32x4;
typedef __attribute__((ext_vector_type(8)))  int i32x8;
typedef __attribute__((ext_vector_type(4)))  int i32x4;

#define BB 2
#define CI 192
#define HH 192
#define WW 192
#define HW 36864
#define C3 64

#if defined(__HIP_DEVICE_COMPILE__) && __has_builtin(__builtin_amdgcn_tensor_load_to_lds)
#define HAVE_TDM 1
#else
#define HAVE_TDM 0
#endif

__device__ __forceinline__ v8f wmma_f16(v16h a, v16h b, v8f c) {
  // D = A(16x32 f16) * B(32x16 f16) + C(16x16 f32)
  return __builtin_amdgcn_wmma_f32_16x16x32_f16(false, a, false, b, (short)0, c,
                                                false, false);
}

// Load a 16-half WMMA operand fragment from a [row][K] layout: 8 contiguous
// halves at p (K = kg..kg+7) and 8 at p+16 (K = kg+16..kg+23).
__device__ __forceinline__ v16h ldfrag(const _Float16* p) {
  v8h lo = *reinterpret_cast<const v8h*>(p);
  v8h hi = *reinterpret_cast<const v8h*>(p + 16);
  v16h r;
#pragma unroll
  for (int i = 0; i < 8; ++i) { r[i] = lo[i]; r[i + 8] = hi[i]; }
  return r;
}

#if HAVE_TDM
typedef __attribute__((address_space(3))) const void lds_cvoid;
__device__ __forceinline__ unsigned int lds_addr32(const void* p) {
  return (unsigned int)(uintptr_t)(lds_cvoid*)p;
}
// Issue a TDM load of a 2D tile (f16 elements) into LDS. 1D = tile1/td1 == 1.
// D# per CDNA5 ISA 8.3/8.4; groups 2/3 zero (dims >2 unused).
__device__ __forceinline__ void tdm_load(unsigned int lds_off, const void* gaddr,
                                         unsigned int tile0, unsigned int tile1,
                                         unsigned int td0, unsigned int td1,
                                         unsigned long long stride0) {
  unsigned long long ga = (unsigned long long)(uintptr_t)gaddr;
  u32x4 g0;
  g0[0] = 1u;                                   // count=1 (valid), no gather
  g0[1] = lds_off;                              // lds_addr (bytes)
  g0[2] = (unsigned int)ga;                     // global_addr[31:0]
  g0[3] = (unsigned int)((ga >> 32) & 0x1FFFFFFu) | 0x80000000u;  // [56:32] | type=2
  i32x8 g1;
  g1[0] = 0x00010000;                           // data_size = 2 bytes, mask=0
  g1[1] = (int)((td0 & 0xFFFFu) << 16);         // tensor_dim0[15:0]
  g1[2] = (int)((td0 >> 16) | ((td1 & 0xFFFFu) << 16));  // td0[31:16] | td1[15:0]
  g1[3] = (int)((td1 >> 16) | ((tile0 & 0xFFFFu) << 16));// td1[31:16] | tile_dim0
  g1[4] = (int)(tile1 & 0xFFFFu);               // tile_dim1 (tile_dim2 = 0)
  g1[5] = (int)(unsigned int)stride0;           // tensor_dim0_stride[31:0]
  g1[6] = (int)((stride0 >> 32) & 0xFFFFu);     // stride0[47:32]
  g1[7] = 0;
  i32x4 z4 = {0, 0, 0, 0};
#if __clang_major__ >= 23
  i32x8 z8 = {0, 0, 0, 0, 0, 0, 0, 0};
  __builtin_amdgcn_tensor_load_to_lds(g0, g1, z4, z4, z8, 0);
#else
  __builtin_amdgcn_tensor_load_to_lds(g0, g1, z4, z4, 0);
#endif
}
#endif  // HAVE_TDM

// ---------------------------------------------------------------- LN stats
__global__ void k_ln_stats(const float* __restrict__ xl, const float* __restrict__ xr,
                           float* __restrict__ mul, float* __restrict__ rsl,
                           float* __restrict__ mur, float* __restrict__ rsr) {
  int idx = blockIdx.x * blockDim.x + threadIdx.x;
  if (idx >= 2 * BB * HW) return;
  int side = idx / (BB * HW);
  int rem  = idx % (BB * HW);
  int b = rem / HW, p = rem % HW;
  const float* base = (side ? xr : xl) + (size_t)b * CI * HW + p;
  float s = 0.f, s2 = 0.f;
  for (int c = 0; c < CI; ++c) { float v = base[(size_t)c * HW]; s += v; s2 += v * v; }
  float m = s * (1.0f / CI);
  float var = s2 * (1.0f / CI) - m * m;
  float r = rsqrtf(var + 1e-6f);
  if (side) { mur[rem] = m; rsr[rem] = r; } else { mul[rem] = m; rsl[rem] = r; }
}

// ------------------------------------------------- 1x1 conv GEMM via WMMA
// Y[o,p] = sum_c W[o,c] * LN?(X[c,p]); block: 12 waves, block tile 192x32
// (each wave one M-tile x two N-tiles), K=192 in 6 steps of 32.
// isQ: outA = [b][w][h][c64] (height Q slices), outB = [b][h][w][c64] (width Q).
// !isQ: outA = [b][c64][w][h] (height V, A-operand layout).
__global__ __launch_bounds__(384)
void k_conv(const float* __restrict__ Wm, const float* __restrict__ X,
            const float* __restrict__ lnw, const float* __restrict__ lnb,
            const float* __restrict__ mu, const float* __restrict__ rstd,
            int isQ, _Float16* __restrict__ out,
            _Float16* __restrict__ outA, _Float16* __restrict__ outB) {
  const int b   = blockIdx.y;
  const int p0  = blockIdx.x * 32;
  const int tid = threadIdx.x, lane = tid & 31, wave = tid >> 5;
  const float* Xb = X + (size_t)b * CI * HW;
  __shared__ _Float16 xT[32][32];   // [n][k]: B-fragments contiguous in K
  v8f acc0 = {0.f, 0.f, 0.f, 0.f, 0.f, 0.f, 0.f, 0.f};
  v8f acc1 = acc0;
  const int mrow = wave * 16 + (lane & 15);
  const int kg   = (lane >> 4) * 8;
  for (int kk = 0; kk < 6; ++kk) {
    for (int s = tid; s < 1024; s += 384) {
      int k = s >> 5, n = s & 31;
      int c = kk * 32 + k, p = p0 + n;
      float v = Xb[(size_t)c * HW + p];
      if (isQ) {
        float m_ = mu[b * HW + p], r_ = rstd[b * HW + p];
        v = (v - m_) * r_ * lnw[c] + lnb[c];
      }
      xT[n][k] = (_Float16)v;
      if (kk < 5) __builtin_prefetch(&Xb[(size_t)(c + 32) * HW + p], 0, 1);
    }
    __syncthreads();
    const float* wp = Wm + (size_t)mrow * CI + kk * 32 + kg;
    v16h af;
#pragma unroll
    for (int i = 0; i < 8; ++i) { af[i] = (_Float16)wp[i]; af[i + 8] = (_Float16)wp[16 + i]; }
    v16h bf0 = ldfrag(&xT[lane & 15][kg]);
    v16h bf1 = ldfrag(&xT[16 + (lane & 15)][kg]);
    acc0 = wmma_f16(af, bf0, acc0);
    acc1 = wmma_f16(af, bf1, acc1);
    __syncthreads();
  }
  const int r0 = wave * 16 + (lane >> 4) * 8;
#pragma unroll
  for (int sub = 0; sub < 2; ++sub) {
    const int p = p0 + sub * 16 + (lane & 15);
    const int w = p % WW, h = p / WW;
#pragma unroll
    for (int r = 0; r < 8; ++r) {
      int m = r0 + r;
      float fv = sub ? acc1[r] : acc0[r];
      _Float16 v = (_Float16)fv;
      out[(size_t)(b * CI + m) * HW + p] = v;
      if (isQ) {
        if (m < C3)          outA[(((size_t)b * WW + w) * HH + h) * C3 + m] = v;
        else if (m < 2 * C3) outB[(((size_t)b * HH + h) * WW + w) * C3 + (m - C3)] = v;
      } else if (m < C3) {
        outA[((size_t)(b * C3 + m) * WW + w) * HH + h] = v;
      }
    }
  }
}

// ------------------------------------- axial attention scores + softmax
// Q/K stored per-slice contiguous as [slice][row][c64]; S = scale*Q K^T,
// P = row-softmax(S) (192x192, K=64). S in registers (16 waves x 9 tiles).
__global__ __launch_bounds__(512)
void k_attn_score(const _Float16* __restrict__ q0, const _Float16* __restrict__ k0,
                  const float* __restrict__ scp,
                  _Float16* __restrict__ P, _Float16* __restrict__ PT, int writeP) {
  const int slice = blockIdx.x, b = blockIdx.y;
  const int tid = threadIdx.x, lane = tid & 31, wave = tid >> 5;
  const _Float16* qp = q0 + ((size_t)b * 192 + slice) * (192 * C3);
  const _Float16* kp = k0 + ((size_t)b * 192 + slice) * (192 * C3);
  __shared__ _Float16 Qs[192][C3];  // [row][k]
  __shared__ _Float16 Ks[192][C3];  // [col][k]
  __shared__ float red[192][12];
  __shared__ float rowv[192];
#if HAVE_TDM
  if (wave == 0) {
    tdm_load(lds_addr32(&Qs[0][0]), qp, 192 * C3, 1, 192 * C3, 1, 192 * C3);
    tdm_load(lds_addr32(&Ks[0][0]), kp, 192 * C3, 1, 192 * C3, 1, 192 * C3);
    __builtin_amdgcn_s_wait_tensorcnt(0);
  }
#else
  {
    _Float16* Qf = &Qs[0][0];
    _Float16* Kf = &Ks[0][0];
    for (int s = tid; s < 192 * C3; s += 512) { Qf[s] = qp[s]; Kf[s] = kp[s]; }
  }
#endif
  __syncthreads();
  const float sc = *scp;
  const int kg = (lane >> 4) * 8;
  v8f acc[9];
#pragma unroll
  for (int j = 0; j < 9; ++j) {
    int t = wave * 9 + j, mt = t / 12, nt = t % 12;
    v8f a = {0.f, 0.f, 0.f, 0.f, 0.f, 0.f, 0.f, 0.f};
#pragma unroll
    for (int kk = 0; kk < 2; ++kk) {
      v16h af = ldfrag(&Qs[mt * 16 + (lane & 15)][kk * 32 + kg]);
      v16h bf = ldfrag(&Ks[nt * 16 + (lane & 15)][kk * 32 + kg]);
      a = wmma_f16(af, bf, a);
    }
    float part[8];
#pragma unroll
    for (int r = 0; r < 8; ++r) {
      a[r] *= sc;
      float m = a[r];
      for (int off = 1; off < 16; off <<= 1) m = fmaxf(m, __shfl_xor(m, off));
      part[r] = m;
    }
    acc[j] = a;
    if ((lane & 15) == 0)
#pragma unroll
      for (int r = 0; r < 8; ++r) red[mt * 16 + kg + r][nt] = part[r];
  }
  __syncthreads();
  if (tid < 192) {
    float m = red[tid][0];
    for (int i = 1; i < 12; ++i) m = fmaxf(m, red[tid][i]);
    rowv[tid] = m;
  }
  __syncthreads();
#pragma unroll
  for (int j = 0; j < 9; ++j) {
    int t = wave * 9 + j, mt = t / 12, nt = t % 12;
    float part[8];
#pragma unroll
    for (int r = 0; r < 8; ++r) {
      int row = mt * 16 + kg + r;
      acc[j][r] = __expf(acc[j][r] - rowv[row]);
      float s = acc[j][r];
      for (int off = 1; off < 16; off <<= 1) s += __shfl_xor(s, off);
      part[r] = s;
    }
    if ((lane & 15) == 0)
#pragma unroll
      for (int r = 0; r < 8; ++r) red[mt * 16 + kg + r][nt] = part[r];
  }
  __syncthreads();
  if (tid < 192) {
    float s = 0.f;
    for (int i = 0; i < 12; ++i) s += red[tid][i];
    rowv[tid] = s;
  }
  __syncthreads();
  const size_t pbase = (size_t)(b * 192 + slice) * 192 * 192;
#pragma unroll
  for (int j = 0; j < 9; ++j) {
    int t = wave * 9 + j, mt = t / 12, nt = t % 12;
    int col = nt * 16 + (lane & 15);
    int r0 = mt * 16 + kg;
    v8h pk;
#pragma unroll
    for (int r = 0; r < 8; ++r) {
      float v = acc[j][r] / rowv[r0 + r];
      pk[r] = (_Float16)v;
      if (writeP) P[pbase + (size_t)(r0 + r) * 192 + col] = (_Float16)v;
    }
    *reinterpret_cast<v8h*>(PT + pbase + (size_t)col * 192 + r0) = pk;  // transposed
  }
}

// --------------------------------------- apply attention + residual + ls
// Two phases: out[c,n] = ls[c]*sum_k V[c,k]*Bm[k,n] + x;  64x192, K=192.
// V rows contiguous (stride vcs); Bm = [slice][k][n] f16.
__global__ __launch_bounds__(512)
void k_attn_apply(const _Float16* __restrict__ V0, const _Float16* __restrict__ V1,
                  size_t vbs, int vcs,
                  const _Float16* __restrict__ B0, const _Float16* __restrict__ B1,
                  float* __restrict__ out0, float* __restrict__ out1,
                  const float* __restrict__ x0, const float* __restrict__ x1,
                  const float* __restrict__ ls0, const float* __restrict__ ls1,
                  int cb, int axis) {
  const int slice = blockIdx.x, b = blockIdx.y;
  const int tid = threadIdx.x, lane = tid & 31, wave = tid >> 5;
  __shared__ _Float16 Vs[C3][192];   // [c][k]
  __shared__ _Float16 Bt[192][32];   // [n][k] per 32-K chunk
  const int kg = (lane >> 4) * 8;
  for (int ph = 0; ph < 2; ++ph) {
    __syncthreads();
    const _Float16* vp = (ph ? V1 : V0) + (size_t)b * vbs + slice * 192;
    const _Float16* Bp = (ph ? B1 : B0) + (size_t)(b * 192 + slice) * 192 * 192;
    float* op = ph ? out1 : out0;
    const float* xp = ph ? x1 : x0;
    const float* lsp = ph ? ls1 : ls0;
#if HAVE_TDM
    if (wave == 0) {
      tdm_load(lds_addr32(&Vs[0][0]), vp, 192, C3, (unsigned int)vcs, C3,
               (unsigned long long)vcs);
      __builtin_amdgcn_s_wait_tensorcnt(0);
    }
#else
    for (int s = tid; s < C3 * 192; s += 512) {
      int c = s / 192, i = s % 192;
      Vs[c][i] = vp[(size_t)c * vcs + i];
    }
#endif
    v8f acc[3];
#pragma unroll
    for (int j = 0; j < 3; ++j) acc[j] = (v8f){0.f, 0.f, 0.f, 0.f, 0.f, 0.f, 0.f, 0.f};
    for (int kk = 0; kk < 6; ++kk) {
      __syncthreads();
      for (int s = tid; s < 32 * 192; s += 512) {
        int k = s / 192, n = s % 192;
        Bt[n][k] = Bp[(size_t)(kk * 32 + k) * 192 + n];
      }
      __syncthreads();
#pragma unroll
      for (int j = 0; j < 3; ++j) {
        int t = wave * 3 + j, mt = t / 12, nt = t % 12;
        v16h af = ldfrag(&Vs[mt * 16 + (lane & 15)][kk * 32 + kg]);
        v16h bf = ldfrag(&Bt[nt * 16 + (lane & 15)][kg]);
        acc[j] = wmma_f16(af, bf, acc[j]);
      }
    }
#pragma unroll
    for (int j = 0; j < 3; ++j) {
      int t = wave * 3 + j, mt = t / 12, nt = t % 12;
      int n = nt * 16 + (lane & 15);
#pragma unroll
      for (int r = 0; r < 8; ++r) {
        int c = cb + mt * 16 + kg + r;
        size_t ad = (axis == 0)
            ? ((size_t)(b * CI + c) * HH + n) * WW + slice
            : ((size_t)(b * CI + c) * HH + slice) * WW + n;
        op[ad] = acc[j][r] * lsp[c] + xp[ad];
      }
    }
  }
}

// ---------------------------------------------------- channel-SE branch
__global__ void k_pool(const _Float16* __restrict__ ql, const _Float16* __restrict__ qr,
                       float* __restrict__ pool) {
  int b = blockIdx.x / C3, c64 = blockIdx.x % C3;
  size_t base = (size_t)(b * CI + 128 + c64) * HW;
  float s = 0.f;
  for (int p = threadIdx.x; p < HW; p += 256)
    s += (float)ql[base + p] + (float)qr[base + p];
  __shared__ float sm[256];
  sm[threadIdx.x] = s;
  __syncthreads();
  for (int o = 128; o > 0; o >>= 1) {
    if (threadIdx.x < o) sm[threadIdx.x] += sm[threadIdx.x + o];
    __syncthreads();
  }
  if (threadIdx.x == 0) pool[b * C3 + c64] = sm[0] * (1.0f / HW);
}

__global__ void k_se(const float* __restrict__ pool, const float* __restrict__ w1,
                     const float* __restrict__ w2, float* __restrict__ a) {
  __shared__ float h1[BB][8];
  __shared__ float lg[BB][128];
  int tid = threadIdx.x;
  if (tid < 16) {
    int b = tid >> 3, j = tid & 7;
    float s = 0.f;
    for (int i = 0; i < 64; ++i) s += pool[b * 64 + i] * w1[j * 64 + i];
    h1[b][j] = fmaxf(s, 0.f);
  }
  __syncthreads();
  {
    int b = tid >> 7, o = tid & 127;
    float s = 0.f;
    for (int j = 0; j < 8; ++j) s += h1[b][j] * w2[o * 8 + j];
    lg[b][o] = s;
  }
  __syncthreads();
  if (tid < BB) {
    float m = -1e30f;
    for (int o = 0; o < 128; ++o) m = fmaxf(m, lg[tid][o]);
    float s = 0.f;
    for (int o = 0; o < 128; ++o) { lg[tid][o] = __expf(lg[tid][o] - m); s += lg[tid][o]; }
    float inv = 1.0f / s;
    for (int o = 0; o < 128; ++o) a[tid * 128 + o] = lg[tid][o] * inv;
  }
}

__global__ void k_se_apply(const float* __restrict__ a,
                           const _Float16* __restrict__ vl, const _Float16* __restrict__ vr,
                           const float* __restrict__ xl, const float* __restrict__ xr,
                           const float* __restrict__ ls1, const float* __restrict__ ls2,
                           float* __restrict__ out_l, float* __restrict__ out_r) {
  size_t idx = (size_t)blockIdx.x * blockDim.x + threadIdx.x;
  const size_t half = (size_t)BB * C3 * HW;
  if (idx >= 2 * half) return;
  int side = (int)(idx / half);
  size_t rem = idx % half;
  int b = (int)(rem / ((size_t)C3 * HW));
  size_t r2 = rem % ((size_t)C3 * HW);
  int c64 = (int)(r2 / HW);
  int p = (int)(r2 % HW);
  int c = 128 + c64;
  float av = a[b * 128 + side * 64 + c64];
  const _Float16* v = side ? vr : vl;
  const float* x = side ? xr : xl;
  const float* ls = side ? ls2 : ls1;
  float* o = side ? out_r : out_l;
  size_t ad = (size_t)(b * CI + c) * HW + p;
  o[ad] = av * (float)v[ad] * ls[c] + x[ad];
}

// ------------------------------------------------------------------ host
extern "C" void kernel_launch(void* const* d_in, const int* in_sizes, int n_in,
                              void* d_out, int out_size, void* d_ws, size_t ws_size,
                              hipStream_t stream) {
  const float* x_l  = (const float*)d_in[0];
  const float* x_r  = (const float*)d_in[1];
  const float* lnlw = (const float*)d_in[2];
  const float* lnlb = (const float*)d_in[3];
  const float* lnrw = (const float*)d_in[4];
  const float* lnrb = (const float*)d_in[5];
  const float* wq_l = (const float*)d_in[6];
  const float* wq_r = (const float*)d_in[7];
  const float* wv_l = (const float*)d_in[8];
  const float* wv_r = (const float*)d_in[9];
  const float* w1   = (const float*)d_in[10];
  const float* w2   = (const float*)d_in[11];
  const float* sch  = (const float*)d_in[12];
  const float* scw  = (const float*)d_in[13];
  const float* ls1  = (const float*)d_in[14];
  const float* ls2  = (const float*)d_in[15];

  float* out_l = (float*)d_out;
  float* out_r = out_l + (size_t)BB * CI * HW;

  char* ws = (char*)d_ws;
  size_t off = 0;
  auto take = [&](size_t bytes) { char* p = ws + off; off += (bytes + 255) & ~(size_t)255; return p; };
  const size_t QV = (size_t)BB * CI * HW;      // half elements
  const size_t QT = (size_t)BB * C3 * HW;
  const size_t PB = (size_t)BB * 192 * 192 * 192;
  const size_t ST = (size_t)BB * HW;           // float elements
  _Float16* q_l  = (_Float16*)take(QV * 2);
  _Float16* q_r  = (_Float16*)take(QV * 2);
  _Float16* v_l  = (_Float16*)take(QV * 2);
  _Float16* v_r  = (_Float16*)take(QV * 2);
  _Float16* qS0l = (_Float16*)take(QT * 2);   // [b][w][h][c64] height Q
  _Float16* qS0r = (_Float16*)take(QT * 2);
  _Float16* qS1l = (_Float16*)take(QT * 2);   // [b][h][w][c64] width Q
  _Float16* qS1r = (_Float16*)take(QT * 2);
  _Float16* vTl  = (_Float16*)take(QT * 2);   // [b][c64][w][h] height V
  _Float16* vTr  = (_Float16*)take(QT * 2);
  _Float16* buf1 = (_Float16*)take(PB * 2);
  _Float16* buf2 = (_Float16*)take(PB * 2);
  float* mul  = (float*)take(ST * 4);
  float* rsl  = (float*)take(ST * 4);
  float* mur  = (float*)take(ST * 4);
  float* rsr  = (float*)take(ST * 4);
  float* pool = (float*)take(BB * C3 * 4);
  float* avec = (float*)take(BB * 128 * 4);

  // 1) LayerNorm stats
  k_ln_stats<<<(2 * BB * HW + 255) / 256, 256, 0, stream>>>(x_l, x_r, mul, rsl, mur, rsr);

  // 2) pointwise convs (WMMA GEMM), LN fused for q
  dim3 cg(HW / 32, BB);
  k_conv<<<cg, 384, 0, stream>>>(wq_l, x_l, lnlw, lnlb, mul, rsl, 1, q_l, qS0l, qS1l);
  k_conv<<<cg, 384, 0, stream>>>(wq_r, x_r, lnrw, lnrb, mur, rsr, 1, q_r, qS0r, qS1r);
  k_conv<<<cg, 384, 0, stream>>>(wv_l, x_l, lnlw, lnlb, mul, rsl, 0, v_l, vTl, nullptr);
  k_conv<<<cg, 384, 0, stream>>>(wv_r, x_r, lnrw, lnrb, mur, rsr, 0, v_r, vTr, nullptr);

  dim3 ag(192, BB);
  // 3) height axial attention: slices over w, channels [0,64)
  k_attn_score<<<ag, 512, 0, stream>>>(qS0l, qS0r, sch, buf1, buf2, 1);
  k_attn_apply<<<ag, 512, 0, stream>>>(vTr, vTl, (size_t)C3 * HW, HW, buf2, buf1,
                                       out_l, out_r, x_l, x_r, ls1, ls2, 0, 0);

  // 4) width axial attention: slices over h, channels [64,128), two softmaxes
  k_attn_score<<<ag, 512, 0, stream>>>(qS1l, qS1r, scw, nullptr, buf1, 0);
  k_attn_score<<<ag, 512, 0, stream>>>(qS1r, qS1l, scw, nullptr, buf2, 0);
  k_attn_apply<<<ag, 512, 0, stream>>>(v_r + (size_t)C3 * HW, v_l + (size_t)C3 * HW,
                                       (size_t)CI * HW, HW, buf1, buf2,
                                       out_l, out_r, x_l, x_r, ls1, ls2, C3, 1);

  // 5) channel SE branch, channels [128,192)
  k_pool<<<BB * C3, 256, 0, stream>>>(q_l, q_r, pool);
  k_se<<<1, 256, 0, stream>>>(pool, w1, w2, avec);
  k_se_apply<<<(int)((2 * (size_t)BB * C3 * HW + 255) / 256), 256, 0, stream>>>(
      avec, v_l, v_r, x_l, x_r, ls1, ls2, out_l, out_r);
}
